// GraniteAttention_41034117546415
// MI455X (gfx1250) — matrix operand order repaired and checked by reading, not compiled, not run on hardware
//
#include <hip/hip_runtime.h>
#include <hip/hip_bf16.h>
#include <cstdint>

// ---------------- problem constants (GraniteAttention) ----------------
#define B_      2
#define S_      2048
#define D_      4096
#define H_      32
#define KVH_    8
#define HD_     128
#define GROUPS_ (H_ / KVH_)      // 4
#define SCALING_ 0.08838834764831845f
#define NEG_     (-1000000000.0f)

typedef __attribute__((ext_vector_type(16))) __bf16 v16bf;
typedef __attribute__((ext_vector_type(8)))  float  v8f;

// ---------------------------------------------------------------------
// WMMA fragment layouts used in this file (wave32; contiguous k order for A and B)
//  A (16x32 bf16): lane L -> row M=L%16; K=0..15 (lanes 0-15)
//                  or K=16..31 (lanes 16-31); per-lane reads contiguous.
//  B (32x16 bf16): lane L -> col N=L%16; K=0..15 (lanes 0-15), 16..31 (16-31).
//  C/D (16x16 f32): VGPR r -> M=r (lanes 0-15) / M=r+8 (lanes 16-31), N=L%16.
// B operands are PRE-PACKED fragment-major: [K/32][N/16][32 lanes][16 bf16],
// so a B fragment is one 32-byte contiguous load per lane.
// ---------------------------------------------------------------------

__device__ __forceinline__ v16bf load_frag_A(const __bf16* __restrict__ A, int lda,
                                             int m0, int k0) {
  const int lane = threadIdx.x & 31;
  const int m    = m0 + (lane & 15);
  const int kb   = (lane < 16) ? 0 : 16;
  const __bf16* row = A + (long)m * lda + k0;
  v16bf f;
#pragma unroll
  for (int v = 0; v < 8; ++v) {
    const int k = kb + (v << 1);
    f[2 * v]     = row[k];
    f[2 * v + 1] = row[k + 1];
  }
  return f;
}

__device__ __forceinline__ v16bf load_frag_A_f32(const float* __restrict__ A, int lda,
                                                 int m0, int k0) {
  const int lane = threadIdx.x & 31;
  const int m    = m0 + (lane & 15);
  const int kb   = (lane < 16) ? 0 : 16;
  const float* row = A + (long)m * lda + k0;
  v16bf f;
#pragma unroll
  for (int v = 0; v < 8; ++v) {
    const int k = kb + (v << 1);
    f[2 * v]     = (__bf16)row[k];
    f[2 * v + 1] = (__bf16)row[k + 1];
  }
  return f;
}

// packed-B fragment: one contiguous 32B read per lane
__device__ __forceinline__ v16bf load_frag_B_packed(const __bf16* __restrict__ Bp,
                                                    int NT, int kt, int nt) {
  const int lane = threadIdx.x & 31;
  const __bf16* p = Bp + ((((long)kt * NT + nt) * 32) + lane) * 16;
  return *(const v16bf*)p;
}

// B fragment where logical B = K^T (Km row-major, ldk): per-lane contiguous in k
__device__ __forceinline__ v16bf load_frag_BT(const __bf16* __restrict__ Km, int ldk,
                                              int k0, int n0) {
  const int lane = threadIdx.x & 31;
  const int n    = n0 + (lane & 15);
  const int kb   = (lane < 16) ? 0 : 16;
  const __bf16* row = Km + (long)n * ldk + k0 + kb;
  v16bf f;
#pragma unroll
  for (int v = 0; v < 8; ++v) {
    f[2 * v]     = row[2 * v];
    f[2 * v + 1] = row[2 * v + 1];
  }
  return f;
}

__device__ __forceinline__ v8f wmma_bf16(v16bf a, v16bf b, v8f c) {
  return __builtin_amdgcn_wmma_f32_16x16x32_bf16(false, a, false, b, (short)0, c,
                                                 false, false);
}

// ---------------------------------------------------------------------
// 1) f32 -> bf16 elementwise (hidden states)
// ---------------------------------------------------------------------
__global__ void cvt_f32_to_bf16(const float* __restrict__ src, __bf16* __restrict__ dst,
                                long n) {
  long i = (long)blockIdx.x * blockDim.x + threadIdx.x;
  if (i < n) dst[i] = (__bf16)src[i];
}

// ---------------------------------------------------------------------
// 2) f32 weights [K,N] -> bf16 fragment-major packed [K/32][N/16][32][16]
//    one thread per (kt,nt,lane): 16 strided reads, one 32B vector store
// ---------------------------------------------------------------------
__global__ void cvt_pack_weights(const float* __restrict__ src, __bf16* __restrict__ pk,
                                 int N, int K) {
  long idx = (long)blockIdx.x * blockDim.x + threadIdx.x;
  const int NT = N >> 4;
  const long total = ((long)(K >> 5)) * NT * 32;
  if (idx >= total) return;
  const int lane = (int)(idx & 31);
  long t = idx >> 5;
  const int nt = (int)(t % NT);
  const int kt = (int)(t / NT);
  const int kb = (lane & 16) ? 16 : 0;
  const int n  = nt * 16 + (lane & 15);
  const float* s = src + (long)(kt * 32 + kb) * N + n;
  v16bf f;
#pragma unroll
  for (int e = 0; e < 16; ++e) f[e] = (__bf16)s[(long)e * N];
  *(v16bf*)(pk + idx * 16) = f;
}

// bf16 [B,S,KVH*HD] (v_lin) -> fragment-major packed V per (b,kvh):
// [B][KVH][S/32][HD/16][32][16]
__global__ void pack_v_kernel(const __bf16* __restrict__ vlin, __bf16* __restrict__ pk) {
  long idx = (long)blockIdx.x * blockDim.x + threadIdx.x;
  const int NT = HD_ >> 4;                       // 8
  const long total = (long)B_ * KVH_ * (S_ >> 5) * NT * 32;
  if (idx >= total) return;
  const int lane = (int)(idx & 31);
  long t = idx >> 5;
  const int nt = (int)(t % NT); t /= NT;
  const int kt = (int)(t % (S_ >> 5)); t /= (S_ >> 5);
  const int h  = (int)(t % KVH_);
  const int b  = (int)(t / KVH_);
  const int kb = (lane & 16) ? 16 : 0;
  const int d  = nt * 16 + (lane & 15);
  const __bf16* s = vlin + ((long)b * S_ + kt * 32 + kb) * (KVH_ * HD_) + (long)h * HD_ + d;
  v16bf f;
#pragma unroll
  for (int e = 0; e < 16; ++e) f[e] = s[(long)e * (KVH_ * HD_)];
  *(v16bf*)(pk + idx * 16) = f;
}

// ---------------------------------------------------------------------
// 3) bf16 WMMA GEMM, packed B, async-LDS double-buffered A staging.
//    256 thr = 8 waves; block tile 128x64; wave tile 16x64 (4 accumulators).
// ---------------------------------------------------------------------
#define GEMM_MBLK 128
#define GEMM_NBLK 64

template <typename OutT>
__global__ void __launch_bounds__(256)
gemm_bf16_packedB(const __bf16* __restrict__ A, const __bf16* __restrict__ Bp,
                  OutT* __restrict__ C, int N, int K) {
  __shared__ __bf16 Asmem[2][GEMM_MBLK][32];
  const int tid  = threadIdx.x;
  const int wave = tid >> 5;
  const int lane = tid & 31;
  const int mblk = blockIdx.y * GEMM_MBLK;
  const int n0   = blockIdx.x * GEMM_NBLK;
  const int nt0  = n0 >> 4;
  const int NT   = N >> 4;

  // async stage of one 128x32 A slice (8 KB): 2 x b128 per thread
  auto stage = [&](int buf, int k0) {
    const __bf16* srcbase = A + (long)mblk * K + k0;
#pragma unroll
    for (int i = 0; i < 2; ++i) {
      const int c  = tid * 2 + i;
      const int r  = c >> 2;
      const int ch = c & 3;
      const __bf16* g = srcbase + (long)r * K + ch * 8;
      unsigned loff = (unsigned)(uintptr_t)&Asmem[buf][r][ch * 8];
      asm volatile("global_load_async_to_lds_b128 %0, %1, off"
                   :: "v"(loff), "v"(g) : "memory");
    }
  };

  v8f acc[4] = {};
  stage(0, 0);
  int buf = 0;
  for (int k0 = 0; k0 < K; k0 += 32, buf ^= 1) {
    asm volatile("s_wait_asynccnt 0x0" ::: "memory");
    __syncthreads();
    if (k0 + 32 < K) stage(buf ^ 1, k0 + 32);

    // A fragment from LDS: two contiguous 16B chunks per lane (ds_load_b128)
    const int m  = wave * 16 + (lane & 15);
    const int kb = (lane < 16) ? 0 : 16;
    const __bf16* row = &Asmem[buf][m][0];
    v16bf a;
#pragma unroll
    for (int v = 0; v < 8; ++v) {
      const int k = kb + (v << 1);
      a[2 * v]     = row[k];
      a[2 * v + 1] = row[k + 1];
    }
    const int kt = k0 >> 5;
#pragma unroll
    for (int t = 0; t < 4; ++t) {
      v16bf b = load_frag_B_packed(Bp, NT, kt, nt0 + t);
      acc[t] = wmma_bf16(a, b, acc[t]);
    }
  }

  const int n  = lane & 15;
  const int mo = (lane < 16) ? 0 : 8;
#pragma unroll
  for (int t = 0; t < 4; ++t)
#pragma unroll
    for (int r = 0; r < 8; ++r)
      C[(long)(mblk + wave * 16 + mo + r) * N + n0 + t * 16 + n] = (OutT)acc[t][r];
}

// ---------------------------------------------------------------------
// 4) RoPE + head transpose: [B,S,NH*HD] bf16 -> [B,NH,S,HD] bf16
// ---------------------------------------------------------------------
__global__ void rope_kernel(const __bf16* __restrict__ lin, const float* __restrict__ cosb,
                            const float* __restrict__ sinb, __bf16* __restrict__ out,
                            int NH, long total) {
  long idx = (long)blockIdx.x * blockDim.x + threadIdx.x;
  if (idx >= total) return;
  const int half = HD_ / 2;
  int  d = (int)(idx % half);  long t = idx / half;
  int  s = (int)(t % S_);      t /= S_;
  int  h = (int)(t % NH);      int b = (int)(t / NH);
  const __bf16* src = lin + ((long)b * S_ + s) * ((long)NH * HD_) + (long)h * HD_;
  const long cs = ((long)b * S_ + s) * HD_;
  float x1 = (float)src[d];
  float x2 = (float)src[d + half];
  float o1 = x1 * cosb[cs + d]        - x2 * sinb[cs + d];
  float o2 = x2 * cosb[cs + d + half] + x1 * sinb[cs + d + half];
  __bf16* dst = out + (((long)b * NH + h) * S_ + s) * HD_;
  dst[d]        = (__bf16)o1;
  dst[d + half] = (__bf16)o2;
}

// ---------------------------------------------------------------------
// 5) scores: W = scale*(Q K^T) + causal mask (raw). 8 waves/block, one
//    q-tile per block; Q fragments (full HD) held in registers; fully
//    masked k-tiles skip WMMA and store NEG directly.
// ---------------------------------------------------------------------
__global__ void __launch_bounds__(256)
scores_kernel(const __bf16* __restrict__ Q, const __bf16* __restrict__ Kr,
              float* __restrict__ W) {
  const int wave = threadIdx.x >> 5;
  const int lane = threadIdx.x & 31;
  const int bh   = blockIdx.y;
  const int b    = bh / H_;
  const int h    = bh % H_;
  const int kvh  = h / GROUPS_;
  const int q0   = blockIdx.x * 16;
  const __bf16* Qp = Q  + ((long)b * H_   + h)   * S_ * HD_;
  const __bf16* Kp = Kr + ((long)b * KVH_ + kvh) * S_ * HD_;

  v16bf qf[4];
#pragma unroll
  for (int kd = 0; kd < 4; ++kd) qf[kd] = load_frag_A(Qp, HD_, q0, kd * 32);

  const int n  = lane & 15;
  const int mo = (lane < 16) ? 0 : 8;
  float* Wq0 = W + ((long)bh * S_ + q0) * S_;

  for (int kt = wave; kt < S_ / 16; kt += 8) {
    const int k0   = kt * 16;
    const int kcol = k0 + n;
    if (k0 <= q0 + 15) {             // tile intersects the causal region
      v8f acc = {};
#pragma unroll
      for (int kd = 0; kd < 4; ++kd) {
        v16bf bt = load_frag_BT(Kp, HD_, kd * 32, k0);
        acc = wmma_bf16(qf[kd], bt, acc);
      }
#pragma unroll
      for (int r = 0; r < 8; ++r) {
        const int qrow = mo + r;     // row within tile
        const float m  = (kcol <= q0 + qrow) ? 0.0f : NEG_;
        Wq0[(long)qrow * S_ + kcol] = acc[r] * SCALING_ + m;
      }
    } else {                          // fully masked: no WMMA needed
#pragma unroll
      for (int r = 0; r < 8; ++r)
        Wq0[(long)(mo + r) * S_ + kcol] = NEG_;
    }
  }
}

// ---------------------------------------------------------------------
// 6) in-place row softmax (attn_weights is a reference output)
// ---------------------------------------------------------------------
__global__ void softmax_kernel(float* __restrict__ W) {
  float* p = W + (long)blockIdx.x * S_;
  __shared__ float red[256];
  const int tid = threadIdx.x;
  float m = -3.0e38f;
  for (int i = tid; i < S_; i += 256) m = fmaxf(m, p[i]);
  red[tid] = m; __syncthreads();
  for (int s = 128; s > 0; s >>= 1) {
    if (tid < s) red[tid] = fmaxf(red[tid], red[tid + s]);
    __syncthreads();
  }
  m = red[0]; __syncthreads();
  float sum = 0.0f;
  for (int i = tid; i < S_; i += 256) {
    float e = __expf(p[i] - m);
    p[i] = e;
    sum += e;
  }
  red[tid] = sum; __syncthreads();
  for (int s = 128; s > 0; s >>= 1) {
    if (tid < s) red[tid] += red[tid + s];
    __syncthreads();
  }
  const float inv = 1.0f / red[0];
  for (int i = tid; i < S_; i += 256) p[i] *= inv;
}

// ---------------------------------------------------------------------
// 7) PV: ctx[b,q,h*HD+d] = sum_k P * V.  2 waves/block, wave tile 16x64,
//    packed V fragments, k-loop limited to the causal extent (P==0 beyond).
// ---------------------------------------------------------------------
__global__ void __launch_bounds__(64)
pv_kernel(const float* __restrict__ W, const __bf16* __restrict__ Vpk,
          __bf16* __restrict__ ctx) {
  const int wave = threadIdx.x >> 5;
  const int lane = threadIdx.x & 31;
  const int bh   = blockIdx.y;
  const int b    = bh / H_;
  const int h    = bh % H_;
  const int kvh  = h / GROUPS_;
  const int q0   = blockIdx.x * 16;
  const float*  Pp = W + (long)bh * S_ * S_;
  const __bf16* Vp = Vpk + ((long)b * KVH_ + kvh) * S_ * HD_;   // packed per (b,kvh)

  const int ktmax = (q0 + 16 + 31) >> 5;     // slices with any P != 0
  v8f acc[4] = {};
  for (int kt = 0; kt < ktmax; ++kt) {
    v16bf a = load_frag_A_f32(Pp, S_, q0, kt * 32);
#pragma unroll
    for (int t = 0; t < 4; ++t) {
      v16bf v = load_frag_B_packed(Vp, HD_ >> 4, kt, wave * 4 + t);
      acc[t] = wmma_bf16(a, v, acc[t]);
    }
  }
  const int n  = lane & 15;
  const int mo = (lane < 16) ? 0 : 8;
#pragma unroll
  for (int t = 0; t < 4; ++t) {
    const int d = wave * 64 + t * 16 + n;
#pragma unroll
    for (int r = 0; r < 8; ++r)
      ctx[((long)b * S_ + q0 + mo + r) * (H_ * HD_) + (long)h * HD_ + d] =
          (__bf16)acc[t][r];
  }
}

// ---------------------------------------------------------------------
// launcher
// ---------------------------------------------------------------------
extern "C" void kernel_launch(void* const* d_in, const int* in_sizes, int n_in,
                              void* d_out, int out_size, void* d_ws, size_t ws_size,
                              hipStream_t stream) {
  (void)in_sizes; (void)n_in; (void)out_size; (void)ws_size;
  const float* hs   = (const float*)d_in[0];   // (B,S,D)
  const float* cosb = (const float*)d_in[1];   // (B,S,HD)
  const float* sinb = (const float*)d_in[2];   // (B,S,HD)
  // d_in[3] attention_mask: causal, computed analytically (skips a 1 GB read)
  const float* Wq = (const float*)d_in[4];     // (D, H*HD)
  const float* Wk = (const float*)d_in[5];     // (D, KVH*HD)
  const float* Wv = (const float*)d_in[6];     // (D, KVH*HD)
  const float* Wo = (const float*)d_in[7];     // (H*HD, D)

  float* attn_out = (float*)d_out;                       // B*S*D f32
  float* attn_w   = attn_out + (long)B_ * S_ * D_;       // B*H*S*S f32

  const long N_HS = (long)B_ * S_ * D_;        // 16,777,216
  const long N_WQ = (long)D_ * H_ * HD_;       // 16,777,216
  const long N_WK = (long)D_ * KVH_ * HD_;     //  4,194,304
  __bf16* w = (__bf16*)d_ws;
  long off = 0;
  __bf16* hsb  = w + off; off += N_HS;         // hidden bf16 (row-major)
  __bf16* wqb  = w + off; off += N_WQ;         // packed Wq
  __bf16* wkb  = w + off; off += N_WK;         // packed Wk
  __bf16* wvb  = w + off; off += N_WK;         // packed Wv
  __bf16* wob  = w + off; off += N_WQ;         // packed Wo
  __bf16* qlin = w + off; off += N_HS;         // (B,S,H*HD)
  __bf16* klin = w + off; off += N_WK;         // (B,S,KVH*HD)
  __bf16* vlin = w + off; off += N_WK;         // (B,S,KVH*HD)
  // aliases (producers fully consumed before these are written)
  __bf16* Qr  = hsb;          // (B,H,S,HD)
  __bf16* Kr  = wqb;          // (B,KVH,S,HD)
  __bf16* Vpk = wqb + N_WK;   // packed V, fits inside old Wq region
  __bf16* ctx = qlin;         // (B,S,H*HD)

  const int CT = 256;
  // 1) conversions / weight packing
  cvt_f32_to_bf16<<<dim3((N_HS + CT - 1) / CT), CT, 0, stream>>>(hs, hsb, N_HS);
  {
    long tw = N_WQ / 16;   // (K/32)*(N/16)*32 threads
    cvt_pack_weights<<<dim3((tw + CT - 1) / CT), CT, 0, stream>>>(Wq, wqb, H_ * HD_, D_);
    long tk = N_WK / 16;
    cvt_pack_weights<<<dim3((tk + CT - 1) / CT), CT, 0, stream>>>(Wk, wkb, KVH_ * HD_, D_);
    cvt_pack_weights<<<dim3((tk + CT - 1) / CT), CT, 0, stream>>>(Wv, wvb, KVH_ * HD_, D_);
    cvt_pack_weights<<<dim3((tw + CT - 1) / CT), CT, 0, stream>>>(Wo, wob, D_, H_ * HD_);
  }

  // 2) QKV projections (M = B*S = 4096)
  const int M = B_ * S_;
  gemm_bf16_packedB<__bf16><<<dim3((H_ * HD_) / GEMM_NBLK, M / GEMM_MBLK), 256, 0, stream>>>(
      hsb, wqb, qlin, H_ * HD_, D_);
  gemm_bf16_packedB<__bf16><<<dim3((KVH_ * HD_) / GEMM_NBLK, M / GEMM_MBLK), 256, 0, stream>>>(
      hsb, wkb, klin, KVH_ * HD_, D_);
  gemm_bf16_packedB<__bf16><<<dim3((KVH_ * HD_) / GEMM_NBLK, M / GEMM_MBLK), 256, 0, stream>>>(
      hsb, wvb, vlin, KVH_ * HD_, D_);

  // 3) RoPE (Q,K) + V fragment packing
  {
    long tq = (long)B_ * H_ * S_ * (HD_ / 2);
    rope_kernel<<<dim3((tq + CT - 1) / CT), CT, 0, stream>>>(qlin, cosb, sinb, Qr, H_, tq);
    long tk = (long)B_ * KVH_ * S_ * (HD_ / 2);
    rope_kernel<<<dim3((tk + CT - 1) / CT), CT, 0, stream>>>(klin, cosb, sinb, Kr, KVH_, tk);
    long tv = ((long)B_ * KVH_ * S_ * HD_) / 16;
    pack_v_kernel<<<dim3((tv + CT - 1) / CT), CT, 0, stream>>>(vlin, Vpk);
  }

  // 4) raw masked scores -> attn_weights region
  scores_kernel<<<dim3(S_ / 16, B_ * H_), 256, 0, stream>>>(Qr, Kr, attn_w);

  // 5) softmax in place
  softmax_kernel<<<dim3(B_ * H_ * S_), 256, 0, stream>>>(attn_w);

  // 6) P @ V -> context (bf16)
  pv_kernel<<<dim3(S_ / 16, B_ * H_), 64, 0, stream>>>(attn_w, Vpk, ctx);

  // 7) output projection -> attn_output (f32)
  gemm_bf16_packedB<float><<<dim3(D_ / GEMM_NBLK, M / GEMM_MBLK), 256, 0, stream>>>(
      ctx, wob, attn_out, D_, H_ * HD_);
}
